// DAGERC_27075473834503
// MI455X (gfx1250) — matrix operand order repaired and checked by reading, not compiled, not run on hardware
//
#include <hip/hip_runtime.h>

#define USE_ASYNC 1   // stage GEMM tiles with global_load_async_to_lds_b128 (ASYNCcnt)

typedef __bf16 bf16;
typedef __attribute__((ext_vector_type(16))) __bf16 bf16x16;
typedef __attribute__((ext_vector_type(8)))  float   f32x8;

// Problem constants
static constexpr int CB = 128;     // batch
static constexpr int CN = 128;     // nodes
static constexpr int CM = 9;       // speakers
static constexpr int CD = 1024;    // input dim
static constexpr int CH = 256;     // hidden
static constexpr int CL = 2;       // layers
static constexpr int CC = 7;       // classes
static constexpr int CIN = 1792;   // H*(L+1)+D

// ---------------------------------------------------------------------------
// WMMA bf16 GEMM:  C[m,n] = act( sum_k A[m,k] * W[n,k] + bias[n] )
// A: (Md,Kd) bf16, row stride lda.  W: (Nd,Kd) bf16 (i.e. C = A @ W^T).
// CONTRACT: Md % 64 == 0, Nd % 128 == 0, Kd % 32 == 0 (all call sites comply)
// -> no row/col guards, async copies issue with full EXEC.
// 64x128 tile per 256-thread block (8 wave32s). Wave w -> rows 16*(w&3),
// cols 64*(w>>2); each wave: 1 A-frag x 4 B-frags -> 4 back-to-back v_wmma
// per K-step. Double-buffered LDS staging via async copies (ASYNCcnt),
// one barrier per K-step.  sched_barriers keep the issue order:
//   [async copies of next tile] [10x ds_load frags] [4x v_wmma]
// ---------------------------------------------------------------------------
__global__ __launch_bounds__(256) void gemm_bf16_wmma(
    const bf16* __restrict__ A, long lda,
    const bf16* __restrict__ W,
    const float* __restrict__ bias,
    float* __restrict__ Cf, long ldc,
    bf16* __restrict__ Cb, long ldcb,
    int Md, int Nd, int Kd, int relu)
{
  __shared__ bf16 As[2][64][40];    // +8 pad: 80B row stride keeps 16B alignment
  __shared__ bf16 Bs[2][128][40];

  const int tid  = threadIdx.x;
  const int wave = tid >> 5;
  const int lane = tid & 31;
  const int bm = blockIdx.y * 64;
  const int bn = blockIdx.x * 128;
  const int wm = (wave & 3) * 16;
  const int wn = (wave >> 2) * 64;

  f32x8 acc[4];
#pragma unroll
  for (int j = 0; j < 4; ++j) acc[j] = f32x8{0.f,0.f,0.f,0.f,0.f,0.f,0.f,0.f};

  // staging coordinates: 16B per transfer, full tiles (no guards)
  const int ar = tid >> 2;          // A row 0..63
  const int ac = (tid & 3) << 3;    // A col 0,8,16,24
  const int br0 = tid >> 2;         // B rows 0..63  (j=0)
  const int br1 = br0 + 64;         // B rows 64..127 (j=1)

  // fragment addressing per ISA 16-bit wave32 layouts
  const int mrow  = wm + (lane & 15);
  const int abase = (lane < 16) ? 0 : 8;    // A: low lanes K0..7/16..23, high K8..15/24..31
  const int kb    = (lane < 16) ? 0 : 8;    // B: low lanes K0..7/16..23, high K8..15/24..31
  const int nbase = wn + (lane & 15);

  auto stage = [&](int buf, int k0) {
#if USE_ASYNC
    const bf16* ga = A + (long)(bm + ar) * lda + k0 + ac;
    asm volatile("global_load_async_to_lds_b128 %0, %1, off"
                 :: "v"((unsigned)(uintptr_t)&As[buf][ar][ac]), "v"(ga)
                 : "memory");
    const bf16* gb0 = W + (long)(bn + br0) * (long)Kd + k0 + ac;
    asm volatile("global_load_async_to_lds_b128 %0, %1, off"
                 :: "v"((unsigned)(uintptr_t)&Bs[buf][br0][ac]), "v"(gb0)
                 : "memory");
    const bf16* gb1 = W + (long)(bn + br1) * (long)Kd + k0 + ac;
    asm volatile("global_load_async_to_lds_b128 %0, %1, off"
                 :: "v"((unsigned)(uintptr_t)&Bs[buf][br1][ac]), "v"(gb1)
                 : "memory");
#else
    uint4 av = *(const uint4*)(A + (long)(bm + ar) * lda + k0 + ac);
    *(uint4*)&As[buf][ar][ac] = av;
    uint4 bv0 = *(const uint4*)(W + (long)(bn + br0) * (long)Kd + k0 + ac);
    *(uint4*)&Bs[buf][br0][ac] = bv0;
    uint4 bv1 = *(const uint4*)(W + (long)(bn + br1) * (long)Kd + k0 + ac);
    *(uint4*)&Bs[buf][br1][ac] = bv1;
#endif
  };

  stage(0, 0);
#if USE_ASYNC
  asm volatile("s_wait_asynccnt 0x0" ::: "memory");
#endif
  __syncthreads();

  const int nsteps = Kd >> 5;
  for (int s = 0; s < nsteps; ++s) {
    const int cur = s & 1;
    if (s + 1 < nsteps) stage(cur ^ 1, (s + 1) << 5);
    __builtin_amdgcn_sched_barrier(0);   // keep next-tile copies ahead of frag reads

    // load ALL fragments first (distinct registers), then 4 back-to-back WMMAs
    union Frag { bf16x16 v; uint4 q[2]; };
    Frag fa, fb[4];
    fa.q[0] = *(const uint4*)&As[cur][mrow][abase];
    fa.q[1] = *(const uint4*)&As[cur][mrow][16 + abase];
#pragma unroll
    for (int j = 0; j < 4; ++j) {
      fb[j].q[0] = *(const uint4*)&Bs[cur][nbase + 16 * j][kb];
      fb[j].q[1] = *(const uint4*)&Bs[cur][nbase + 16 * j][16 + kb];
    }
    __builtin_amdgcn_sched_barrier(0);   // all ds_loads issued before any WMMA
#pragma unroll
    for (int j = 0; j < 4; ++j) {
      acc[j] = __builtin_amdgcn_wmma_f32_16x16x32_bf16(false, fa.v, false, fb[j].v,
                                                       (short)0, acc[j], false, false);
    }
    __builtin_amdgcn_sched_barrier(0);

#if USE_ASYNC
    if (s + 1 < nsteps) asm volatile("s_wait_asynccnt 0x0" ::: "memory");
#endif
    __syncthreads();
  }

  // C/D layout: lanes 0-15 -> M=r, lanes 16-31 -> M=r+8; N = lane%16
  const int rb = (lane < 16) ? 0 : 8;
#pragma unroll
  for (int j = 0; j < 4; ++j) {
    const int col = bn + wn + 16 * j + (lane & 15);
    const float bv = bias ? bias[col] : 0.f;
#pragma unroll
    for (int r = 0; r < 8; ++r) {
      const int grow = bm + wm + rb + r;
      float v = acc[j][r] + bv;
      if (relu) v = fmaxf(v, 0.f);
      if (Cf) Cf[(long)grow * ldc  + col] = v;
      if (Cb) Cb[(long)grow * ldcb + col] = (bf16)v;
    }
  }
}

// ---------------------------------------------------------------------------
// Elementwise / small kernels
// ---------------------------------------------------------------------------
__global__ void cast_f32_bf16(const float* __restrict__ s, bf16* __restrict__ d, long n) {
  long i = (long)blockIdx.x * blockDim.x + threadIdx.x;
  if (i < n) d[i] = (bf16)s[i];
}

__global__ void cast2d_f32_bf16(const float* __restrict__ s, long ss,
                                bf16* __restrict__ d, long ds, long rows, long cols) {
  long i = (long)blockIdx.x * blockDim.x + threadIdx.x;
  if (i >= rows * cols) return;
  long r = i / cols, c = i - r * cols;
  d[r * ds + c] = (bf16)s[r * ss + c];
}

// K[b,n,:] = b2 ; H1f = 0
__global__ void init_kh(float* __restrict__ Kf, const float* __restrict__ b2,
                        float* __restrict__ H1f) {
  long i = (long)blockIdx.x * blockDim.x + threadIdx.x;
  if (i >= (long)CB * CN * CH) return;
  int h = (int)(i & (CH - 1));
  Kf[i]  = b2[h];
  H1f[i] = 0.f;
}

// Mp[b,n,h] = sum_m s_adj[b,m,n] * P[b,m,h]
__global__ void mp_kernel(const float* __restrict__ sadj, const float* __restrict__ P,
                          float* __restrict__ Mp) {
  long i = (long)blockIdx.x * blockDim.x + threadIdx.x;
  if (i >= (long)CB * CN * CH) return;
  int h = (int)(i & (CH - 1));
  long t = i >> 8;
  int n = (int)(t & (CN - 1));
  int b = (int)(t >> 7);
  const float* sa = sadj + (long)b * CM * CN + n;
  const float* Pr = P + (long)b * CM * CH + h;
  float acc = 0.f;
#pragma unroll
  for (int m = 0; m < CM; ++m) acc += sa[m * CN] * Pr[m * CH];
  Mp[i] = acc;
}

// Pnext[b,m,h] = sum_n s_adj[b,m,n] * H1[b,n,h]
__global__ void pnext_kernel(const float* __restrict__ sadj, const float* __restrict__ H1f,
                             float* __restrict__ Pf, bf16* __restrict__ Pb) {
  long i = (long)blockIdx.x * blockDim.x + threadIdx.x;
  if (i >= (long)CB * CM * CH) return;
  int h = (int)(i & (CH - 1));
  long t = i >> 8;
  int m = (int)(t % CM);
  int b = (int)(t / CM);
  const float* sa = sadj + ((long)b * CM + m) * CN;
  const float* Hb = H1f + (long)b * CN * CH + h;
  float acc = 0.f;
  for (int n = 0; n < CN; ++n) acc += sa[n] * Hb[(long)n * CH];
  Pf[i] = acc;
  Pb[i] = (bf16)acc;
}

// GRUCell combine: gi/gh are (R,3H); h' = (1-z)*tanh(in + r*hn) + z*hprev
__global__ void gru_combine(const float* __restrict__ gi, long ldgi,
                            const float* __restrict__ gh, long ldgh,
                            const float* __restrict__ ghb,     // bias fallback when gh==null
                            const float* __restrict__ hprev, long ldh,
                            float* __restrict__ outf, long ldof,
                            bf16* __restrict__ outb, long ldob, int R) {
  long i = (long)blockIdx.x * blockDim.x + threadIdx.x;
  if (i >= (long)R * CH) return;
  int r = (int)(i >> 8);
  int c = (int)(i & (CH - 1));
  const float* gir = gi + (long)r * ldgi;
  float ir = gir[c], iz = gir[c + CH], in_ = gir[c + 2 * CH];
  float hr, hz, hn;
  if (gh) {
    const float* ghr = gh + (long)r * ldgh;
    hr = ghr[c]; hz = ghr[c + CH]; hn = ghr[c + 2 * CH];
  } else {
    hr = ghb[c]; hz = ghb[c + CH]; hn = ghb[c + 2 * CH];
  }
  float rg = 1.f / (1.f + __expf(-(ir + hr)));
  float zg = 1.f / (1.f + __expf(-(iz + hz)));
  float ng = tanhf(in_ + rg * hn);
  float hp = hprev ? hprev[(long)r * ldh + c] : 0.f;
  float o  = (1.f - zg) * ng + zg * hp;
  if (outf) outf[(long)r * ldof + c] = o;
  if (outb) outb[(long)r * ldob + c] = (bf16)o;
}

// Per-step masked attention. One block per batch b.
// alpha[n] = K[b,n,:].Q[b,:] - (1 - adj[b,i,n]*(n<i))*1e30 ; softmax ; Msum = attn@H1
// Mcat[b, 0:H) = Msum ; Mcat[b, H:2H) = Mp[b,i,:]
__global__ void attn_kernel(const float* __restrict__ Kf, const float* __restrict__ Qf,
                            const float* __restrict__ H1f, const float* __restrict__ adj,
                            const float* __restrict__ Mpf, bf16* __restrict__ Mcat, int i) {
  const int b = blockIdx.x;
  const int t = threadIdx.x;
  __shared__ float sal[CN];
  __shared__ float red[CN];

  if (t < CN) {
    const float* kr = Kf + ((long)b * CN + t) * CH;
    const float* q  = Qf + (long)b * CH;
    float acc = 0.f;
    for (int h = 0; h < CH; ++h) acc += kr[h] * q[h];
    float valid = (t < i) ? 1.f : 0.f;
    float a = adj[((long)b * CN + i) * CN + t];
    float al = acc - (1.f - a * valid) * 1e30f;
    sal[t] = al;
    red[t] = al;
  }
  __syncthreads();
  for (int s = CN / 2; s > 0; s >>= 1) {
    if (t < s) red[t] = fmaxf(red[t], red[t + s]);
    __syncthreads();
  }
  float mx = red[0];
  __syncthreads();
  if (t < CN) {
    float e = __expf(sal[t] - mx);
    sal[t] = e;
    red[t] = e;
  }
  __syncthreads();
  for (int s = CN / 2; s > 0; s >>= 1) {
    if (t < s) red[t] += red[t + s];
    __syncthreads();
  }
  float sum = red[0];
  __syncthreads();
  if (t < CN) sal[t] = sal[t] / sum;
  __syncthreads();

  const int h = t;   // 256 threads -> one h each
  float acc = 0.f;
  const float* Hb = H1f + (long)b * CN * CH + h;
  for (int n = 0; n < CN; ++n) acc += sal[n] * Hb[(long)n * CH];
  Mcat[(long)b * (2 * CH) + h]      = (bf16)acc;
  Mcat[(long)b * (2 * CH) + CH + h] = (bf16)Mpf[((long)b * CN + i) * CH + h];
}

// Final 7-class head: out[row,c] = h[row,:] . wf[c,:] + bf[c]
__global__ void head_kernel(const float* __restrict__ hf, const float* __restrict__ wf,
                            const float* __restrict__ bfv, float* __restrict__ out) {
  long tid = (long)blockIdx.x * blockDim.x + threadIdx.x;
  if (tid >= (long)CB * CN * 8) return;
  int c = (int)(tid & 7);
  long row = tid >> 3;
  if (c >= CC) return;
  const float* hr = hf + row * CH;
  const float* wr = wf + (long)c * CH;
  float acc = bfv[c];
  for (int k = 0; k < CH; ++k) acc += hr[k] * wr[k];
  out[row * CC + c] = acc;
}

// ---------------------------------------------------------------------------
// Host orchestration
// ---------------------------------------------------------------------------
extern "C" void kernel_launch(void* const* d_in, const int* in_sizes, int n_in,
                              void* d_out, int out_size, void* d_ws, size_t ws_size,
                              hipStream_t stream) {
  const float* x    = (const float*)d_in[0];
  const float* adj  = (const float*)d_in[1];
  // d_in[2] = s_mask (unused by the math)
  const float* sfeat = (const float*)d_in[3];
  const float* sadj  = (const float*)d_in[4];
  const float* fc1_w = (const float*)d_in[5];
  const float* fc1_b = (const float*)d_in[6];
  const float* fc2_w = (const float*)d_in[7];
  const float* fc2_b = (const float*)d_in[8];
  const float* gat_w1 = (const float*)d_in[9];
  const float* gat_b1 = (const float*)d_in[10];
  const float* gat_w2 = (const float*)d_in[11];
  const float* gat_b2 = (const float*)d_in[12];
  const float* gru_wih = (const float*)d_in[13];
  const float* gru_whh = (const float*)d_in[14];
  const float* gru_bih = (const float*)d_in[15];
  const float* gru_bhh = (const float*)d_in[16];
  const float* lin_w = (const float*)d_in[17];
  const float* lin_b = (const float*)d_in[18];
  const float* mlp_w0 = (const float*)d_in[19];
  const float* mlp_b0 = (const float*)d_in[20];
  const float* mlp_wh = (const float*)d_in[21];
  const float* mlp_bh = (const float*)d_in[22];
  const float* mlp_wf = (const float*)d_in[23];
  const float* mlp_bf = (const float*)d_in[24];

  const long BN = (long)CB * CN;        // 16384
  const long BM = (long)CB * CM;        // 1152
  const long HCATLD = CIN;              // 1792
  const long NH = (long)CN * CH;        // 32768
  const long NHC = (long)CN * CIN;      // 229376

  // --- workspace bump allocator (256B aligned) ---
  char* ws = (char*)d_ws;
  size_t off = 0;
  auto alloc = [&](size_t bytes) -> void* {
    void* p = ws + off;
    off += (bytes + 255) & ~(size_t)255;
    return p;
  };
  bf16* Hcat   = (bf16*)alloc((size_t)BN * CIN * 2);
  bf16* sfb    = (bf16*)alloc((size_t)BM * CD * 2);
  bf16* w_fc1  = (bf16*)alloc((size_t)CH * CD * 2);
  bf16* w_fc2  = (bf16*)alloc((size_t)CH * CD * 2);
  bf16* w_gat1 = (bf16*)alloc((size_t)CL * CH * CH * 2);
  bf16* w_gat2 = (bf16*)alloc((size_t)CL * CH * CH * 2);
  bf16* w_wih  = (bf16*)alloc((size_t)CL * 3 * CH * CH * 2);
  bf16* w_whh  = (bf16*)alloc((size_t)CL * 3 * CH * CH * 2);
  bf16* w_lin  = (bf16*)alloc((size_t)CL * CH * 2 * CH * 2);
  bf16* w_mlp0 = (bf16*)alloc((size_t)CH * CIN * 2);
  bf16* w_mlph = (bf16*)alloc((size_t)(CL - 1) * CH * CH * 2);
  float* H1f   = (float*)alloc((size_t)BN * CH * 4);
  float* Kf    = (float*)alloc((size_t)BN * CH * 4);   // reused as mlp hidden fp32 at the end
  float* Mpf   = (float*)alloc((size_t)BN * CH * 4);
  float* Pcur_f  = (float*)alloc((size_t)BM * CH * 4);
  bf16*  Pcur_b  = (bf16*)alloc((size_t)BM * CH * 2);
  float* Pnext_f = (float*)alloc((size_t)BM * CH * 4);
  bf16*  Pnext_b = (bf16*)alloc((size_t)BM * CH * 2);
  float* gi = (float*)alloc((size_t)BM * 3 * CH * 4);
  float* gh = (float*)alloc((size_t)BM * 3 * CH * 4);
  float* Qf = (float*)alloc((size_t)CB * CH * 4);
  bf16*  Mcat   = (bf16*)alloc((size_t)CB * 2 * CH * 2);
  float* Mfus_f = (float*)alloc((size_t)CB * CH * 4);
  bf16*  Mfus_b = (bf16*)alloc((size_t)CB * CH * 2);
  bf16*  mh_bf  = (bf16*)alloc((size_t)BN * CH * 2);
  if (off > ws_size) return;  // workspace too small: bail safely

  auto castN = [&](const float* s, bf16* d, long n) {
    cast_f32_bf16<<<dim3((unsigned)((n + 255) / 256)), 256, 0, stream>>>(s, d, n);
  };
  auto gemm = [&](const bf16* A, long lda, const bf16* W, const float* bias,
                  float* Cf, long ldc, bf16* Cb, long ldcb,
                  int Md, int Nd, int Kd, int relu) {
    dim3 grid((unsigned)((Nd + 127) / 128), (unsigned)((Md + 63) / 64));
    gemm_bf16_wmma<<<grid, 256, 0, stream>>>(A, lda, W, bias, Cf, ldc, Cb, ldcb,
                                             Md, Nd, Kd, relu);
  };

  // --- weight / input downcasts to bf16 ---
  castN(fc1_w, w_fc1, (long)CH * CD);
  castN(fc2_w, w_fc2, (long)CH * CD);
  castN(gat_w1, w_gat1, (long)CL * CH * CH);
  castN(gat_w2, w_gat2, (long)CL * CH * CH);
  castN(gru_wih, w_wih, (long)CL * 3 * CH * CH);
  castN(gru_whh, w_whh, (long)CL * 3 * CH * CH);
  castN(lin_w, w_lin, (long)CL * CH * 2 * CH);
  castN(mlp_w0, w_mlp0, (long)CH * CIN);
  castN(mlp_wh, w_mlph, (long)(CL - 1) * CH * CH);
  castN(sfeat, sfb, BM * CD);
  // x -> Hcat[:, 3H : 3H+D]
  cast2d_f32_bf16<<<dim3((unsigned)((BN * CD + 255) / 256)), 256, 0, stream>>>(
      x, CD, Hcat + (long)(CL + 1) * CH, HCATLD, BN, CD);

  // --- fc2: P0 = relu(s_feature @ fc2_w.T + b) ---
  gemm(sfb, CD, w_fc2, fc2_b, Pcur_f, CH, Pcur_b, CH, (int)BM, CH, CD, 1);
  // --- fc1: H0 = relu(x @ fc1_w.T + b) -> Hcat slice 0 (bf16) ---
  gemm(Hcat + (long)(CL + 1) * CH, HCATLD, w_fc1, fc1_b,
       nullptr, 0, Hcat, HCATLD, (int)BN, CH, CD, 1);

  const unsigned gBNH = (unsigned)((BN * CH + 255) / 256);
  const unsigned gBMH = (unsigned)((BM * CH + 255) / 256);

  for (int l = 0; l < CL; ++l) {
    const bf16* wih_b = w_wih + (size_t)l * 3 * CH * CH;
    const bf16* whh_b = w_whh + (size_t)l * 3 * CH * CH;
    const bf16* w1_b  = w_gat1 + (size_t)l * CH * CH;
    const bf16* w2_b  = w_gat2 + (size_t)l * CH * CH;
    const bf16* lw_b  = w_lin + (size_t)l * CH * 2 * CH;
    const float* bih = gru_bih + (size_t)l * 3 * CH;
    const float* bhh = gru_bhh + (size_t)l * 3 * CH;
    const float* b1  = gat_b1 + (size_t)l * CH;
    const float* b2  = gat_b2 + (size_t)l * CH;
    const float* lb  = lin_b + (size_t)l * CH;
    bf16* Hcur = Hcat + (size_t)l * CH;        // H_list[l] slice, row stride CIN
    bf16* Hout = Hcat + (size_t)(l + 1) * CH;  // H_list[l+1] slice

    // Mp = s_adj^T @ P_cur  (B,N,H)
    mp_kernel<<<gBNH, 256, 0, stream>>>(sadj, Pcur_f, Mpf);
    // K <- broadcast b2 ; H1 <- 0
    init_kh<<<gBNH, 256, 0, stream>>>(Kf, b2, H1f);

    // h0 = GRU(H_cur[:,0], 0)
    gemm(Hcur, NHC, wih_b, bih, gi, 3 * CH, nullptr, 0, CB, 3 * CH, CH, 0);
    gru_combine<<<(CB * CH + 255) / 256, 256, 0, stream>>>(
        gi, 3 * CH, nullptr, 0, bhh, nullptr, 0,
        H1f, NH, Hout, NHC, CB);
    // K[:,0] = h0 @ w2.T + b2
    gemm(Hout, NHC, w2_b, b2, Kf, NH, nullptr, 0, CB, CH, CH, 0);

    for (int i = 1; i < CN; ++i) {
      const bf16* xi = Hcur + (long)i * HCATLD;  // H_cur[:,i], row stride N*CIN
      // Q = x_i @ w1.T + b1
      gemm(xi, NHC, w1_b, b1, Qf, CH, nullptr, 0, CB, CH, CH, 0);
      // masked softmax attention -> Mcat = [Msum | Mp[:,i]]
      attn_kernel<<<CB, 256, 0, stream>>>(Kf, Qf, H1f, adj, Mpf, Mcat, i);
      // Mfused = Mcat @ lw.T + lb
      gemm(Mcat, 2 * CH, lw_b, lb, Mfus_f, CH, Mfus_b, CH, CB, CH, 2 * CH, 0);
      // GRU gates
      gemm(xi, NHC, wih_b, bih, gi, 3 * CH, nullptr, 0, CB, 3 * CH, CH, 0);
      gemm(Mfus_b, CH, whh_b, bhh, gh, 3 * CH, nullptr, 0, CB, 3 * CH, CH, 0);
      // h_i -> H1f[:,i] (fp32) and Hout[:,i] (bf16)
      gru_combine<<<(CB * CH + 255) / 256, 256, 0, stream>>>(
          gi, 3 * CH, gh, 3 * CH, bhh, Mfus_f, CH,
          H1f + (long)i * CH, NH, Hout + (long)i * HCATLD, NHC, CB);
      // K[:,i] = h_i @ w2.T + b2
      gemm(Hout + (long)i * HCATLD, NHC, w2_b, b2,
           Kf + (long)i * CH, NH, nullptr, 0, CB, CH, CH, 0);
    }

    if (l < CL - 1) {
      // P_next_in = s_adj @ H1 ; P_cur = GRU(P_cur, P_next_in)
      pnext_kernel<<<gBMH, 256, 0, stream>>>(sadj, H1f, Pnext_f, Pnext_b);
      gemm(Pcur_b, CH, wih_b, bih, gi, 3 * CH, nullptr, 0, (int)BM, 3 * CH, CH, 0);
      gemm(Pnext_b, CH, whh_b, bhh, gh, 3 * CH, nullptr, 0, (int)BM, 3 * CH, CH, 0);
      gru_combine<<<(unsigned)((BM * CH + 255) / 256), 256, 0, stream>>>(
          gi, 3 * CH, gh, 3 * CH, bhh, Pnext_f, CH,
          Pcur_f, CH, Pcur_b, CH, (int)BM);
    }
  }

  // --- MLP head over Hcat (B*N, 1792) ---
  gemm(Hcat, HCATLD, w_mlp0, mlp_b0, nullptr, 0, mh_bf, CH, (int)BN, CH, CIN, 1);
  gemm(mh_bf, CH, w_mlph, mlp_bh, Kf, CH, nullptr, 0, (int)BN, CH, CH, 1);
  head_kernel<<<(unsigned)((BN * 8 + 255) / 256), 256, 0, stream>>>(
      Kf, mlp_wf, mlp_bf, (float*)d_out);
}
